// ISMAR2023_17961553232369
// MI455X (gfx1250) — matrix operand order repaired and checked by reading, not compiled, not run on hardware
//
#include <hip/hip_runtime.h>
#include <hip/hip_bf16.h>

typedef __attribute__((ext_vector_type(16))) __bf16 v16bf;
typedef __attribute__((ext_vector_type(8)))  __bf16 v8bf;
typedef __attribute__((ext_vector_type(8)))  float  v8f;

__device__ __forceinline__ __bf16 f2bf(float f) {
    unsigned u = __builtin_bit_cast(unsigned, f);
    u += 0x7fffu + ((u >> 16) & 1u);           // round-to-nearest-even
    return __builtin_bit_cast(__bf16, (unsigned short)(u >> 16));
}
__device__ __forceinline__ float fsigmoid(float x) { return 1.0f / (1.0f + __expf(-x)); }
__device__ __forceinline__ float ftanh(float x)    { float e = __expf(-2.0f * x); return (1.0f - e) / (1.0f + e); }
__device__ __forceinline__ float felu(float x)     { return x > 0.0f ? x : __expf(x) - 1.0f; }

// A fragment (contiguous k order k = 16h + s, same as B): lane needs one k-run [p, p+16)
__device__ __forceinline__ v16bf load_afrag(const __bf16* p) {
    v8bf a0 = *(const v8bf*)p;
    v8bf a1 = *(const v8bf*)(p + 8);
    return __builtin_shufflevector(a0, a1, 0,1,2,3,4,5,6,7,8,9,10,11,12,13,14,15);
}
// B fragment (contiguous k order k = 16h + s, same as A, B stored transposed [n][k]): one run [p, p+16)
__device__ __forceinline__ v16bf load_bfrag(const __bf16* p) {
    v8bf b0 = *(const v8bf*)p;
    v8bf b1 = *(const v8bf*)(p + 8);
    return __builtin_shufflevector(b0, b1, 0,1,2,3,4,5,6,7,8,9,10,11,12,13,14,15);
}

// ---------------------------------------------------------------------------
// Kernel 1: gate GRUs (2 layers, hidden=8, input=3) over W=128 steps + softmax
// grid 2 (Local / Remote), block 512 = 64 batches x 8 gate units
// ---------------------------------------------------------------------------
__global__ void __launch_bounds__(512) gate_gru_kernel(
    const float* __restrict__ Local, const float* __restrict__ Remote,
    const float* __restrict__ Wih0, const float* __restrict__ Whh0,
    const float* __restrict__ bih0, const float* __restrict__ bhh0,
    const float* __restrict__ Wih1, const float* __restrict__ Whh1,
    const float* __restrict__ bih1, const float* __restrict__ bhh1,
    float* __restrict__ Lo, float* __restrict__ Ro)
{
    const int s = blockIdx.x;
    const float* src = s ? Remote : Local;
    float* dst = s ? Ro : Lo;
    const int b = threadIdx.x >> 3;   // 0..63
    const int g = threadIdx.x & 7;    // 0..7

    __shared__ float h0[64][8], h1[64][8], eb[64][8], xb[64][3];
    __shared__ float sWih0[72], sWhh0[192], sWih1[192], sWhh1[192];
    __shared__ float sb[4][24];
    for (int i = threadIdx.x; i < 72; i += 512) sWih0[i] = Wih0[i];
    for (int i = threadIdx.x; i < 192; i += 512) {
        sWhh0[i] = Whh0[i]; sWih1[i] = Wih1[i]; sWhh1[i] = Whh1[i];
    }
    if (threadIdx.x < 24) {
        sb[0][threadIdx.x] = bih0[threadIdx.x]; sb[1][threadIdx.x] = bhh0[threadIdx.x];
        sb[2][threadIdx.x] = bih1[threadIdx.x]; sb[3][threadIdx.x] = bhh1[threadIdx.x];
    }
    h0[b][g] = 0.f; h1[b][g] = 0.f;
    __syncthreads();

    for (int t = 0; t < 128; ++t) {
        if (g < 3) xb[b][g] = src[(b * 128 + t) * 132 + 128 + g];
        __syncthreads();
        // layer 0 (input 3)
        float gir = sb[0][g], giz = sb[0][8 + g], gin = sb[0][16 + g];
        #pragma unroll
        for (int i = 0; i < 3; ++i) {
            float x = xb[b][i];
            gir += sWih0[g * 3 + i] * x;
            giz += sWih0[(8 + g) * 3 + i] * x;
            gin += sWih0[(16 + g) * 3 + i] * x;
        }
        float ghr = sb[1][g], ghz = sb[1][8 + g], ghn = sb[1][16 + g];
        #pragma unroll
        for (int i = 0; i < 8; ++i) {
            float hv = h0[b][i];
            ghr += sWhh0[g * 8 + i] * hv;
            ghz += sWhh0[(8 + g) * 8 + i] * hv;
            ghn += sWhh0[(16 + g) * 8 + i] * hv;
        }
        float r = fsigmoid(gir + ghr), z = fsigmoid(giz + ghz);
        float n = ftanh(gin + r * ghn);
        float hn0 = (1.f - z) * n + z * h0[b][g];
        __syncthreads();
        h0[b][g] = hn0;
        __syncthreads();
        // layer 1 (input 8)
        gir = sb[2][g]; giz = sb[2][8 + g]; gin = sb[2][16 + g];
        ghr = sb[3][g]; ghz = sb[3][8 + g]; ghn = sb[3][16 + g];
        #pragma unroll
        for (int i = 0; i < 8; ++i) {
            float x = h0[b][i], hv = h1[b][i];
            gir += sWih1[g * 8 + i] * x;
            giz += sWih1[(8 + g) * 8 + i] * x;
            gin += sWih1[(16 + g) * 8 + i] * x;
            ghr += sWhh1[g * 8 + i] * hv;
            ghz += sWhh1[(8 + g) * 8 + i] * hv;
            ghn += sWhh1[(16 + g) * 8 + i] * hv;
        }
        r = fsigmoid(gir + ghr); z = fsigmoid(giz + ghz);
        n = ftanh(gin + r * ghn);
        float hn1 = (1.f - z) * n + z * h1[b][g];
        __syncthreads();
        h1[b][g] = hn1; eb[b][g] = __expf(hn1);   // |h|<1 -> softmax w/o max-sub is safe
        __syncthreads();
        float ssum = 0.f;
        #pragma unroll
        for (int i = 0; i < 8; ++i) ssum += eb[b][i];
        dst[(b * 128 + t) * 8 + g] = eb[b][g] / ssum;
    }
}

// ---------------------------------------------------------------------------
// Kernel 2: expert layer-1 GEMM via WMMA bf16.
// X(16384x128)=[LA;RA] @ W1cat(128x128), gate-weighted combine -> H(16384x16)
// + BN partial sums per half. grid 256 (64 rows each), block 256 (8 waves);
// wave = one expert column tile, 4 sequential 16-row tiles, B fragments kept
// in registers. L/R half select is block-uniform (pure SALU).
// ---------------------------------------------------------------------------
#define LSTR 136   // padded LDS row stride (halfwords): 272B = 68 banks -> 4-bank lane skew
__global__ void __launch_bounds__(256) expert1_gemm_kernel(
    const float* __restrict__ Local, const float* __restrict__ Remote,
    const float* __restrict__ ae_w1, const float* __restrict__ ae_b1,
    const float* __restrict__ Lo, const float* __restrict__ Ro,
    float* __restrict__ Hbuf, float* __restrict__ stats)
{
    __shared__ alignas(16) __bf16 BmT[128 * LSTR];  // B transposed: [n][k], n = e*16+hh
    __shared__ alignas(16) __bf16 Am[64 * LSTR];    // A tile: [m][k]
    __shared__ float Hacc[64 * 16];
    __shared__ float omsh[64 * 8];                  // gate tile [m][e]

    const int tid = threadIdx.x;
    const int r0 = blockIdx.x * 64;
    const bool isL = blockIdx.x < 128;              // block-uniform half select
    const float* __restrict__ Xsrc = isL ? Local : Remote;
    const float* __restrict__ Om   = isL ? Lo : Ro;
    const int rb = isL ? r0 : (r0 - 8192);

    for (int idx = tid; idx < 16384; idx += 256) {
        int n = idx >> 7, k = idx & 127;
        BmT[n * LSTR + k] = f2bf(ae_w1[(n >> 4) * 2048 + k * 16 + (n & 15)]); // (8,128,16)
    }
    for (int idx = tid; idx < 8192; idx += 256) {
        int m = idx >> 7, k = idx & 127;
        Am[m * LSTR + k] = f2bf(Xsrc[(rb + m) * 132 + k]);
    }
    for (int idx = tid; idx < 512; idx += 256)
        omsh[idx] = Om[(rb + (idx >> 3)) * 8 + (idx & 7)];
    for (int idx = tid; idx < 1024; idx += 256) Hacc[idx] = 0.f;
    __syncthreads();

    const int w = tid >> 5, lane = tid & 31;        // wave32
    const int mrow = lane & 15, hi = lane >> 4;
    const int e = w;                                // wave == expert
    const float b1v = ae_b1[e * 16 + mrow];         // per-lane constant bias

    v16bf bfr[4];                                   // B fragments, reused across row tiles
    #pragma unroll
    for (int kk = 0; kk < 4; ++kk)
        bfr[kk] = load_bfrag(&BmT[(w * 16 + mrow) * LSTR + kk * 32 + hi * 16]);

    #pragma unroll
    for (int mt = 0; mt < 4; ++mt) {
        v8f acc = (v8f){0.f, 0.f, 0.f, 0.f, 0.f, 0.f, 0.f, 0.f};
        const int ma = mt * 16 + mrow;
        #pragma unroll
        for (int kk = 0; kk < 4; ++kk) {
            v16bf af = load_afrag(&Am[ma * LSTR + kk * 32 + hi * 16]);
            acc = __builtin_amdgcn_wmma_f32_16x16x32_bf16(false, af, false, bfr[kk],
                                                          (short)0, acc, false, false);
        }
        #pragma unroll
        for (int j = 0; j < 8; ++j) {
            int m = mt * 16 + j + hi * 8;           // C/D layout: m = reg + 8*(lane/16)
            atomicAdd(&Hacc[m * 16 + mrow], (acc[j] + b1v) * omsh[m * 8 + e]);
        }
    }
    __syncthreads();

    for (int idx = tid; idx < 1024; idx += 256)
        Hbuf[(r0 + (idx >> 4)) * 16 + (idx & 15)] = Hacc[idx];
    if (tid < 16) {
        float s = 0.f, s2 = 0.f;
        for (int m = 0; m < 64; ++m) { float v = Hacc[m * 16 + tid]; s += v; s2 += v * v; }
        int grp = isL ? 0 : 1;                      // BN stats per _expert_mlp call
        atomicAdd(&stats[grp * 32 + tid], s);
        atomicAdd(&stats[grp * 32 + 16 + tid], s2);
    }
}

// ---------------------------------------------------------------------------
// Kernel 3: BN + ELU + gate fold, then expert layer-2 GEMM via WMMA:
// H2(16384x128) @ W2cat(128x16) -> atomic-add into Z(8192x16) (+ mixed bias)
// grid 128, block 256 (8 waves, one 16-row tile each). hn and gate tiles are
// staged in LDS once (tid-based, barrier-separated), wave-uniform row base
// made scalar via readfirstlane.
// ---------------------------------------------------------------------------
__global__ void __launch_bounds__(256) expert2_gemm_kernel(
    const float* __restrict__ Hbuf, const float* __restrict__ stats,
    const float* __restrict__ Lo, const float* __restrict__ Ro,
    const float* __restrict__ ae_w2, const float* __restrict__ ae_b2,
    const float* __restrict__ ae_g, const float* __restrict__ ae_bt,
    float* __restrict__ Z)
{
    __shared__ alignas(16) __bf16 W2T[16 * LSTR];      // W2cat transposed: [o][k]
    __shared__ alignas(16) __bf16 H2m[8][16 * LSTR];   // per-wave A tile
    __shared__ float smean[2][16], sinv[2][16], sg[16], sbt[16];
    __shared__ float sb2[128];                         // ae_b2 (8,16)
    __shared__ float omt[8][128];                      // gate tiles [w][m*8+e]
    __shared__ float hnt[8][256];                      // BN'd hidden [w][m*16+hh]

    const int tid = threadIdx.x;
    if (tid < 32) {
        int grp = tid >> 4, hh = tid & 15;
        float mean = stats[grp * 32 + hh] * (1.0f / 8192.0f);
        float var  = stats[grp * 32 + 16 + hh] * (1.0f / 8192.0f) - mean * mean;
        smean[grp][hh] = mean;
        sinv[grp][hh]  = __frsqrt_rn(var + 1e-5f);
    }
    if (tid < 16) { sg[tid] = ae_g[tid]; sbt[tid] = ae_bt[tid]; }
    if (tid < 128) sb2[tid] = ae_b2[tid];
    for (int idx = tid; idx < 2048; idx += 256) {
        int o = idx & 15, k = idx >> 4;
        W2T[o * LSTR + k] = f2bf(ae_w2[(k >> 4) * 256 + (k & 15) * 16 + o]); // (8,16,16)
    }
    for (int idx = tid; idx < 1024; idx += 256) {      // gate tiles, all 8 waves
        int w2 = idx >> 7, r = idx & 127;
        int row = (blockIdx.x * 8 + w2) * 16 + (r >> 3);
        omt[w2][r] = (row < 8192) ? Lo[row * 8 + (r & 7)]
                                  : Ro[(row - 8192) * 8 + (r & 7)];
    }
    __syncthreads();

    for (int idx = tid; idx < 2048; idx += 256) {      // BN + ELU once per (m,hh)
        int w2 = idx >> 8, r = idx & 255;
        int m = r >> 4, hh = r & 15;
        int row = (blockIdx.x * 8 + w2) * 16 + m;
        int grp = row >> 13;
        float h = Hbuf[row * 16 + hh];
        hnt[w2][r] = felu(sg[hh] * (h - smean[grp][hh]) * sinv[grp][hh] + sbt[hh]);
    }
    __syncthreads();

    for (int idx = tid; idx < 16384; idx += 256) {     // gate-folded bf16 H2 tiles
        int w2 = idx >> 11, r = idx & 2047;
        int m = r >> 7, k = r & 127;
        H2m[w2][m * LSTR + k] = f2bf(hnt[w2][m * 16 + (k & 15)] * omt[w2][m * 8 + (k >> 4)]);
    }
    __syncthreads();

    const int w = tid >> 5, lane = tid & 31;
    const int mrow = lane & 15, hi = lane >> 4;
    const int r0u = __builtin_amdgcn_readfirstlane((blockIdx.x * 8 + w) * 16);
    const int rb  = (r0u < 8192) ? r0u : (r0u - 8192); // scalar Z row base

    v8f acc = (v8f){0.f, 0.f, 0.f, 0.f, 0.f, 0.f, 0.f, 0.f};
    #pragma unroll
    for (int kk = 0; kk < 4; ++kk) {
        v16bf af  = load_afrag(&H2m[w][mrow * LSTR + kk * 32 + hi * 16]);
        v16bf bfv = load_bfrag(&W2T[mrow * LSTR + kk * 32 + hi * 16]);  // ncol = mrow (N=16)
        acc = __builtin_amdgcn_wmma_f32_16x16x32_bf16(false, af, false, bfv,
                                                      (short)0, acc, false, false);
    }

    #pragma unroll
    for (int j = 0; j < 8; ++j) {
        int m = j + hi * 8, o = mrow;
        float bias = 0.f;
        #pragma unroll
        for (int e = 0; e < 8; ++e) bias += omt[w][m * 8 + e] * sb2[e * 16 + o];
        atomicAdd(&Z[(rb + m) * 16 + o], acc[j] + bias);   // ZL + ZR
    }
}

// ---------------------------------------------------------------------------
// Kernel 4: mid GRUs (2 layers, hidden=16) over Z (64,128,16).
// grid 4, block 256 = 16 batches x 16 hidden units
// ---------------------------------------------------------------------------
__global__ void __launch_bounds__(256) mid_gru_kernel(
    const float* __restrict__ Z,
    const float* __restrict__ Wih0, const float* __restrict__ Whh0,
    const float* __restrict__ bih0, const float* __restrict__ bhh0,
    const float* __restrict__ Wih1, const float* __restrict__ Whh1,
    const float* __restrict__ bih1, const float* __restrict__ bhh1,
    float* __restrict__ Zlast)
{
    __shared__ float sWih0[768], sWhh0[768], sWih1[768], sWhh1[768];
    __shared__ float sb[4][48];
    __shared__ float h0[16][16], h1[16][16], xb[16][16];

    const int tid = threadIdx.x;
    for (int i = tid; i < 768; i += 256) {
        sWih0[i] = Wih0[i]; sWhh0[i] = Whh0[i]; sWih1[i] = Wih1[i]; sWhh1[i] = Whh1[i];
    }
    if (tid < 48) { sb[0][tid] = bih0[tid]; sb[1][tid] = bhh0[tid];
                    sb[2][tid] = bih1[tid]; sb[3][tid] = bhh1[tid]; }
    const int bl = tid >> 4, j = tid & 15;
    const int b = blockIdx.x * 16 + bl;
    h0[bl][j] = 0.f; h1[bl][j] = 0.f;
    __syncthreads();

    for (int t = 0; t < 128; ++t) {
        xb[bl][j] = Z[(b * 128 + t) * 16 + j];
        __syncthreads();
        float gir = sb[0][j], giz = sb[0][16 + j], gin = sb[0][32 + j];
        float ghr = sb[1][j], ghz = sb[1][16 + j], ghn = sb[1][32 + j];
        #pragma unroll
        for (int i = 0; i < 16; ++i) {
            float x = xb[bl][i], hv = h0[bl][i];
            gir += sWih0[j * 16 + i] * x;  giz += sWih0[(16 + j) * 16 + i] * x;  gin += sWih0[(32 + j) * 16 + i] * x;
            ghr += sWhh0[j * 16 + i] * hv; ghz += sWhh0[(16 + j) * 16 + i] * hv; ghn += sWhh0[(32 + j) * 16 + i] * hv;
        }
        float r = fsigmoid(gir + ghr), z = fsigmoid(giz + ghz);
        float n = ftanh(gin + r * ghn);
        float hn0 = (1.f - z) * n + z * h0[bl][j];
        __syncthreads();
        h0[bl][j] = hn0;
        __syncthreads();
        gir = sb[2][j]; giz = sb[2][16 + j]; gin = sb[2][32 + j];
        ghr = sb[3][j]; ghz = sb[3][16 + j]; ghn = sb[3][32 + j];
        #pragma unroll
        for (int i = 0; i < 16; ++i) {
            float x = h0[bl][i], hv = h1[bl][i];
            gir += sWih1[j * 16 + i] * x;  giz += sWih1[(16 + j) * 16 + i] * x;  gin += sWih1[(32 + j) * 16 + i] * x;
            ghr += sWhh1[j * 16 + i] * hv; ghz += sWhh1[(16 + j) * 16 + i] * hv; ghn += sWhh1[(32 + j) * 16 + i] * hv;
        }
        r = fsigmoid(gir + ghr); z = fsigmoid(giz + ghz);
        n = ftanh(gin + r * ghn);
        float hn1 = (1.f - z) * n + z * h1[bl][j];
        __syncthreads();
        h1[bl][j] = hn1;
        __syncthreads();
    }
    Zlast[b * 16 + j] = h1[bl][j];
}

// ---------------------------------------------------------------------------
// Kernel 5: decoder expert MLP (input 17, hidden 16, out 128) + BN over 64.
// single block of 1024
// ---------------------------------------------------------------------------
__global__ void __launch_bounds__(1024) decoder_kernel(
    const float* __restrict__ Zlast, const float* __restrict__ Remote,
    const float* __restrict__ Ro,
    const float* __restrict__ w1, const float* __restrict__ b1,
    const float* __restrict__ w2, const float* __restrict__ b2,
    const float* __restrict__ g, const float* __restrict__ bt,
    float* __restrict__ out)
{
    __shared__ float xbuf[64][17], ombuf[64][8], hbuf[64][16], hnbuf[64][16];
    __shared__ float mean[16], inv[16];
    const int tid = threadIdx.x;

    for (int idx = tid; idx < 64 * 17; idx += 1024) {
        int bb = idx / 17, i = idx % 17;
        xbuf[bb][i] = (i < 16) ? Zlast[bb * 16 + i]
                               : Remote[(bb * 128 + 127) * 132 + 131];  // Scale[:, -1]
    }
    for (int idx = tid; idx < 512; idx += 1024) {
        int bb = idx >> 3, e = idx & 7;
        ombuf[bb][e] = Ro[(bb * 128 + 127) * 8 + e];                    // Ro[:, -1]
    }
    __syncthreads();
    {
        int bb = tid >> 4, hh = tid & 15;
        float acc = 0.f;
        #pragma unroll
        for (int e = 0; e < 8; ++e) {
            float s = b1[e * 16 + hh];
            for (int i = 0; i < 17; ++i) s += xbuf[bb][i] * w1[e * 272 + i * 16 + hh];
            acc += ombuf[bb][e] * s;
        }
        hbuf[bb][hh] = acc;
    }
    __syncthreads();
    if (tid < 16) {
        float s = 0.f, s2 = 0.f;
        for (int bb = 0; bb < 64; ++bb) { float v = hbuf[bb][tid]; s += v; s2 += v * v; }
        float m = s * (1.f / 64.f);
        mean[tid] = m;
        inv[tid]  = __frsqrt_rn(s2 * (1.f / 64.f) - m * m + 1e-5f);
    }
    __syncthreads();
    {
        int bb = tid >> 4, hh = tid & 15;
        hnbuf[bb][hh] = felu(g[hh] * (hbuf[bb][hh] - mean[hh]) * inv[hh] + bt[hh]);
    }
    __syncthreads();
    for (int idx = tid; idx < 8192; idx += 1024) {
        int bb = idx >> 7, o = idx & 127;
        float acc = 0.f;
        #pragma unroll
        for (int e = 0; e < 8; ++e) {
            float s = b2[e * 128 + o];
            for (int hh = 0; hh < 16; ++hh) s += hnbuf[bb][hh] * w2[e * 2048 + hh * 128 + o];
            acc += ombuf[bb][e] * s;
        }
        out[bb * 128 + o] = acc;
    }
}

// ---------------------------------------------------------------------------
extern "C" void kernel_launch(void* const* d_in, const int* in_sizes, int n_in,
                              void* d_out, int out_size, void* d_ws, size_t ws_size,
                              hipStream_t stream)
{
    (void)in_sizes; (void)n_in; (void)out_size; (void)ws_size;
    const float* Local  = (const float*)d_in[0];
    const float* Remote = (const float*)d_in[1];
    const float* gWih0 = (const float*)d_in[2],  *gWhh0 = (const float*)d_in[3];
    const float* gbih0 = (const float*)d_in[4],  *gbhh0 = (const float*)d_in[5];
    const float* gWih1 = (const float*)d_in[6],  *gWhh1 = (const float*)d_in[7];
    const float* gbih1 = (const float*)d_in[8],  *gbhh1 = (const float*)d_in[9];
    const float* mWih0 = (const float*)d_in[10], *mWhh0 = (const float*)d_in[11];
    const float* mbih0 = (const float*)d_in[12], *mbhh0 = (const float*)d_in[13];
    const float* mWih1 = (const float*)d_in[14], *mWhh1 = (const float*)d_in[15];
    const float* mbih1 = (const float*)d_in[16], *mbhh1 = (const float*)d_in[17];
    const float* ae_w1 = (const float*)d_in[18], *ae_b1 = (const float*)d_in[19];
    const float* ae_w2 = (const float*)d_in[20], *ae_b2 = (const float*)d_in[21];
    const float* ae_g  = (const float*)d_in[22], *ae_bt = (const float*)d_in[23];
    const float* md_w1 = (const float*)d_in[24], *md_b1 = (const float*)d_in[25];
    const float* md_w2 = (const float*)d_in[26], *md_b2 = (const float*)d_in[27];
    const float* md_g  = (const float*)d_in[28], *md_bt = (const float*)d_in[29];

    float* ws    = (float*)d_ws;
    float* Lo    = ws;                 // 8192*8
    float* Ro    = ws + 65536;         // 8192*8
    float* Hbuf  = ws + 131072;        // 16384*16
    float* stats = ws + 393216;        // 2*(16+16)
    float* Zb    = ws + 393280;        // 8192*16
    float* Zlast = ws + 524352;        // 64*16

    hipMemsetAsync(stats, 0, 64 * sizeof(float), stream);
    hipMemsetAsync(Zb, 0, 131072 * sizeof(float), stream);

    gate_gru_kernel<<<2, 512, 0, stream>>>(Local, Remote, gWih0, gWhh0, gbih0, gbhh0,
                                           gWih1, gWhh1, gbih1, gbhh1, Lo, Ro);
    expert1_gemm_kernel<<<256, 256, 0, stream>>>(Local, Remote, ae_w1, ae_b1,
                                                 Lo, Ro, Hbuf, stats);
    expert2_gemm_kernel<<<128, 256, 0, stream>>>(Hbuf, stats, Lo, Ro,
                                                 ae_w2, ae_b2, ae_g, ae_bt, Zb);
    mid_gru_kernel<<<4, 256, 0, stream>>>(Zb, mWih0, mWhh0, mbih0, mbhh0,
                                          mWih1, mWhh1, mbih1, mbhh1, Zlast);
    decoder_kernel<<<1, 1024, 0, stream>>>(Zlast, Remote, Ro, md_w1, md_b1,
                                           md_w2, md_b2, md_g, md_bt, (float*)d_out);
}
